// MaxMarginLoss_83846351552676
// MI455X (gfx1250) — compile-verified, matrix operand order checked by reading, not run on hardware
//
#include <hip/hip_runtime.h>
#include <hip/hip_bf16.h>
#include <cstdint>

typedef __attribute__((ext_vector_type(16))) _Float16 v16h;
typedef __attribute__((ext_vector_type(8)))  float    v8f;

#define MARGIN 0.1f
#define EPS_N  1e-8f
#define B_ROWS 1024
#define D_DIM  512
#define C_CLS  100000

#define N_STRIPES 125            // class stripes
#define TILES_PER_STRIPE 50      // 50 tiles * 16 classes = 800 classes / stripe

union Frag16 { v16h h; uint4 u[2]; };
union Pack4  { _Float16 h[4]; uint2 u; };
union Pack8  { _Float16 h[8]; uint4 u; };

// ---------------------------------------------------------------------------
// Row-wise L2 normalize (eps-clamped, torch cosine semantics) + fp32 -> f16.
// One wave per 512-float row.
// ---------------------------------------------------------------------------
__global__ void __launch_bounds__(256)
rownorm_f16_kernel(const float* __restrict__ in, _Float16* __restrict__ out,
                   int nrows) {
    int wave = (blockIdx.x * blockDim.x + threadIdx.x) >> 5;
    int lane = threadIdx.x & 31;
    if (wave >= nrows) return;

    const float* row = in + (size_t)wave * D_DIM;
    float4 v[4];
    float ss = 0.f;
#pragma unroll
    for (int i = 0; i < 4; ++i) {
        v[i] = *(const float4*)(row + (size_t)(lane + 32 * i) * 4);
        ss += v[i].x * v[i].x + v[i].y * v[i].y + v[i].z * v[i].z + v[i].w * v[i].w;
    }
#pragma unroll
    for (int m = 16; m >= 1; m >>= 1) ss += __shfl_xor(ss, m, 32);

    float inv = 1.0f / fmaxf(sqrtf(ss), EPS_N);

    _Float16* orow = out + (size_t)wave * D_DIM;
#pragma unroll
    for (int i = 0; i < 4; ++i) {
        Pack4 p;
        p.h[0] = (_Float16)(v[i].x * inv);
        p.h[1] = (_Float16)(v[i].y * inv);
        p.h[2] = (_Float16)(v[i].z * inv);
        p.h[3] = (_Float16)(v[i].w * inv);
        *(uint2*)(orow + (size_t)(lane + 32 * i) * 4) = p.u;
    }
}

// ---------------------------------------------------------------------------
// target_cos[b] = dot(x16[b], cls16[targets[b]]) with f32 accumulation.
// ---------------------------------------------------------------------------
__global__ void __launch_bounds__(256)
target_cos_kernel(const _Float16* __restrict__ x16,
                  const _Float16* __restrict__ cls16,
                  const int* __restrict__ targets,
                  float* __restrict__ tcos) {
    int wave = (blockIdx.x * blockDim.x + threadIdx.x) >> 5;
    int lane = threadIdx.x & 31;
    if (wave >= B_ROWS) return;

    int t = targets[wave];
    const _Float16* xr = x16 + (size_t)wave * D_DIM;
    const _Float16* cr = cls16 + (size_t)t * D_DIM;

    float acc = 0.f;
#pragma unroll
    for (int i = 0; i < 2; ++i) {
        int e = (lane + 32 * i) * 8;
        Pack8 a, b;
        a.u = *(const uint4*)(xr + e);
        b.u = *(const uint4*)(cr + e);
#pragma unroll
        for (int j = 0; j < 8; ++j)
            acc = fmaf((float)a.h[j], (float)b.h[j], acc);
    }
#pragma unroll
    for (int m = 16; m >= 1; m >>= 1) acc += __shfl_xor(acc, m, 32);
    if (lane == 0) tcos[wave] = acc;
}

// ---------------------------------------------------------------------------
// Fused GEMM + hinge reduction, A-resident / B-streamed version.
//   grid = 125 class stripes x 8 b-chunks (1000 blocks, 8 waves each)
//   - each wave keeps its 16x512 f16 A tile in 128 VGPRs for the whole sweep
//   - 16-class B tiles are double-buffered in LDS via async-to-LDS DMA
//     (global_load_async_to_lds_b128 + s_wait_asynccnt), read with
//     ds_load_b128, consumed by two independent WMMA accumulator chains
//   - hinge epilogue on accumulator VGPRs; one atomicAdd per block
// ---------------------------------------------------------------------------
__global__ void __launch_bounds__(256)
hinge_gemm_kernel(const _Float16* __restrict__ x16,
                  const _Float16* __restrict__ cls16,
                  const float* __restrict__ tcos,
                  float* __restrict__ accum) {
    __shared__ __align__(16) _Float16 sB[2][16 * D_DIM];   // 2 x 16 KB
    __shared__ float ssum;

    const int tid  = threadIdx.x;
    const int wave = tid >> 5;
    const int lane = tid & 31;
    const int stripe = blockIdx.x >> 3;   // 0..124
    const int bchunk = blockIdx.x & 7;    // 0..7

    // wave32 WMMA fragment addressing (ISA 7.12.2)
    const int hi = lane >> 4;
    const int ml = lane & 15;
    const int k0 = hi * 8;

    if (tid == 0) ssum = 0.f;

    // --- A tile resident in registers: rows bchunk*128 + wave*16 + ml ------
    const int brow_base = bchunk * 128 + wave * 16;
    const _Float16* arow = x16 + (size_t)(brow_base + ml) * D_DIM;
    Frag16 A[16];
#pragma unroll
    for (int kk = 0; kk < 16; ++kk) {
        A[kk].u[0] = *(const uint4*)(arow + kk * 32 + k0);
        A[kk].u[1] = *(const uint4*)(arow + kk * 32 + k0 + 16);
    }
    // target-cos for this wave's 8 accumulator rows (M = v + 8*hi)
    float tr[8];
#pragma unroll
    for (int v = 0; v < 8; ++v) tr[v] = tcos[brow_base + 8 * hi + v];

    const size_t cls_base = (size_t)stripe * (TILES_PER_STRIPE * 16) * D_DIM;

    // async copy of one 16 KB class tile into LDS buffer (4 b128 per thread)
    auto issue_copy = [&](int t, int buf) {
        const _Float16* g = cls16 + cls_base + (size_t)t * 16 * D_DIM;
#pragma unroll
        for (int i = 0; i < 4; ++i) {
            unsigned e = (unsigned)(tid + 256 * i) * 8;     // f16 elements
            unsigned lds_off = (unsigned)(size_t)
                (__attribute__((address_space(3))) const _Float16*)&sB[buf][e];
            unsigned long long ga = (unsigned long long)(const void*)(g + e);
            asm volatile("global_load_async_to_lds_b128 %0, %1, off"
                         :: "v"(lds_off), "v"(ga) : "memory");
        }
    };

    issue_copy(0, 0);

    float wsum = 0.f;
    for (int t = 0; t < TILES_PER_STRIPE; ++t) {
        const int buf = t & 1;
        if (t + 1 < TILES_PER_STRIPE) {
            issue_copy(t + 1, buf ^ 1);
            asm volatile("s_wait_asynccnt 0x4" ::: "memory");  // tile t landed
        } else {
            asm volatile("s_wait_asynccnt 0x0" ::: "memory");
        }
        __syncthreads();   // whole tile visible to all waves

        const _Float16* brw = &sB[buf][0] + (size_t)ml * D_DIM;
        v8f acc0 = {}, acc1 = {};   // two independent WMMA chains
#pragma unroll
        for (int kk = 0; kk < 16; kk += 2) {
            Frag16 b0, b1;
            b0.u[0] = *(const uint4*)(brw + kk * 32 + k0);
            b0.u[1] = *(const uint4*)(brw + kk * 32 + k0 + 16);
            b1.u[0] = *(const uint4*)(brw + kk * 32 + 32 + k0);
            b1.u[1] = *(const uint4*)(brw + kk * 32 + 32 + k0 + 16);
            acc0 = __builtin_amdgcn_wmma_f32_16x16x32_f16(
                false, A[kk].h, false, b0.h, (short)0, acc0, false, false);
            acc1 = __builtin_amdgcn_wmma_f32_16x16x32_f16(
                false, A[kk + 1].h, false, b1.h, (short)0, acc1, false, false);
        }
        // hinge epilogue: acc[v] holds (M = v + 8*hi, N = ml)
#pragma unroll
        for (int v = 0; v < 8; ++v)
            wsum += fmaxf(0.f, MARGIN - tr[v] + (acc0[v] + acc1[v]));

        __syncthreads();   // reads done before this buffer is overwritten
    }

#pragma unroll
    for (int m = 16; m >= 1; m >>= 1) wsum += __shfl_xor(wsum, m, 32);
    if (lane == 0) atomicAdd(&ssum, wsum);
    __syncthreads();
    if (tid == 0) atomicAdd(accum, ssum);
}

// ---------------------------------------------------------------------------
__global__ void zero_kernel(float* p) { *p = 0.f; }

__global__ void finalize_kernel(const float* __restrict__ accum,
                                float* __restrict__ out) {
    out[0] = accum[0] * (1.0f / (float)B_ROWS) - MARGIN;
}

// ---------------------------------------------------------------------------
extern "C" void kernel_launch(void* const* d_in, const int* in_sizes, int n_in,
                              void* d_out, int out_size, void* d_ws, size_t ws_size,
                              hipStream_t stream) {
    const float* inputs  = (const float*)d_in[0];   // [1024, 512] f32
    const float* cls     = (const float*)d_in[1];   // [100000, 512] f32
    const int*   targets = (const int*)d_in[2];     // [1024]
    float* out = (float*)d_out;

    char* ws = (char*)d_ws;
    const size_t CLS16_BYTES = (size_t)C_CLS * D_DIM * 2;   // 102,400,000
    const size_t X16_BYTES   = (size_t)B_ROWS * D_DIM * 2;  // 1,048,576
    _Float16* cls16 = (_Float16*)ws;
    _Float16* x16   = (_Float16*)(ws + CLS16_BYTES);
    float*    tcos  = (float*)(ws + CLS16_BYTES + X16_BYTES);
    float*    accum = (float*)(ws + CLS16_BYTES + X16_BYTES + 4096);

    rownorm_f16_kernel<<<C_CLS / 8, 256, 0, stream>>>(cls, cls16, C_CLS);
    rownorm_f16_kernel<<<B_ROWS / 8, 256, 0, stream>>>(inputs, x16, B_ROWS);
    target_cos_kernel<<<B_ROWS / 8, 256, 0, stream>>>(x16, cls16, targets, tcos);
    zero_kernel<<<1, 1, 0, stream>>>(accum);
    hinge_gemm_kernel<<<N_STRIPES * 8, 256, 0, stream>>>(x16, cls16, tcos, accum);
    finalize_kernel<<<1, 1, 0, stream>>>(accum, out);
}
